// ByteMultiHeadSelfAttention_42253888258156
// MI455X (gfx1250) — matrix operand order repaired and checked by reading, not compiled, not run on hardware
//
#include <hip/hip_runtime.h>
#include <cstdint>
#include <cstddef>

// ---------------- problem constants (from reference) ----------------
#define B_   2
#define S_   2048
#define E_   2048
#define H_   16
#define KVH_ 4
#define D_   128
#define KV_  512           // KVH_ * D_
#define M_   (B_ * S_)     // 4096 rows for all projections

typedef __bf16 bf16;
typedef __attribute__((ext_vector_type(16))) __bf16 v16bf;
typedef __attribute__((ext_vector_type(8)))  float  v8f;
typedef __attribute__((ext_vector_type(4)))  unsigned int v4u;
typedef __attribute__((ext_vector_type(8)))  int v8i;
typedef __attribute__((ext_vector_type(4)))  int v4i;

union FragU { v16bf v; unsigned int u[8]; unsigned short h[16]; };

#if __has_builtin(__builtin_amdgcn_tensor_load_to_lds) && \
    __has_builtin(__builtin_amdgcn_s_wait_tensorcnt)
#define USE_TDM 1
#else
#define USE_TDM 0
#endif

// ---------------------------------------------------------------
// TDM: issue one 2D-tile global->LDS DMA (Tensor Data Mover, ISA ch.8).
// D# group0: count=1 | lds_addr[63:32] | global_addr[120:64] | type=2[127:126]
// D# group1: data_size=1(2B)[17:16] | pad ctl [20],[24:22],[31:25] |
//            tensor_dim0 @48, tensor_dim1 @80, tile_dim0 @112, tile_dim1 @128,
//            tensor_dim0_stride @160 (48b)
// pad_interval code: 2^(n+1) DWORDs between pads; pad_amount code: n+1 DWORDs.
// ---------------------------------------------------------------
#if USE_TDM
__device__ static inline void tdm_load_2d(unsigned lds_byte_off, const void* gsrc,
                                          unsigned tile_w, unsigned tile_h,
                                          unsigned row_stride_elems,
                                          unsigned pad_en, unsigned pad_interval,
                                          unsigned pad_amount) {
  unsigned long long ga = (unsigned long long)(uintptr_t)gsrc;
  v4u g0 = { 1u,                                   // count = 1 (valid user D#)
             lds_byte_off,                         // lds_addr
             (unsigned)(ga & 0xffffffffu),
             (unsigned)((ga >> 32) & 0x01ffffffu) | (2u << 30) };  // type=2
  unsigned w0 = (1u << 16)                         // data_size = 1 -> 2-byte elems
              | (pad_en << 20) | (pad_interval << 22) | (pad_amount << 25);
  v8i g1 = { (int)w0,
             (int)((tile_w & 0xffffu) << 16),                                  // tensor_dim0 lo
             (int)(((tile_w >> 16) & 0xffffu) | ((tile_h & 0xffffu) << 16)),   // dim0 hi | dim1 lo
             (int)(((tile_h >> 16) & 0xffffu) | ((tile_w & 0xffffu) << 16)),   // dim1 hi | tile_dim0
             (int)(tile_h & 0xffffu),                                          // tile_dim1 (tile_dim2=0)
             (int)row_stride_elems,                                            // dim0_stride lo32
             0, 0 };
  v4i z4 = {};
#if __clang_major__ >= 23
  v8i z8 = {};
  __builtin_amdgcn_tensor_load_to_lds(g0, g1, z4, z4, z8, 0);
#else
  __builtin_amdgcn_tensor_load_to_lds(g0, g1, z4, z4, 0);
#endif
}
#endif

// ---------------------------------------------------------------
// WMMA wrapper: D = A(16x32 bf16) * B(32x16 bf16) + C(16x16 f32)
// ---------------------------------------------------------------
__device__ static inline v8f wmma_bf16(v16bf a, v16bf b, v8f c) {
  return __builtin_amdgcn_wmma_f32_16x16x32_bf16(
      false, a, false, b, (short)0, c, false, false);
}

// A-fragment (16x32, MxK) from a row-major LDS tile (ISA 7.12.2 layout).
__device__ static inline v16bf load_frag_a(const bf16* base, int ld) {
  const int lane  = threadIdx.x & 31;
  const int r     = lane & 15;
  const int khalf = (lane >> 4) << 3;
  FragU f;
#pragma unroll
  for (int g = 0; g < 8; ++g) {
    int k = ((g < 4) ? (2 * g) : (16 + 2 * (g - 4))) + khalf;
    f.u[g] = *(const unsigned int*)(base + r * ld + k);
  }
  return f.v;
}

// B-fragment (32x16, KxN): value(k,n) = base[k*ld + n]; lane n=lane%16, elements over k (same order as A).
__device__ static inline v16bf load_frag_b(const bf16* base, int ld) {
  const int lane  = threadIdx.x & 31;
  const int n     = lane & 15;
  const int khalf = (lane >> 4) << 3;
  FragU f;
#pragma unroll
  for (int g = 0; g < 8; ++g) {
    int k = ((g < 4) ? (2 * g) : (16 + 2 * (g - 4))) + khalf;
    unsigned int lo = *(const unsigned short*)(base + k * ld + n);
    unsigned int hi = *(const unsigned short*)(base + (k + 1) * ld + n);
    f.u[g] = lo | (hi << 16);
  }
  return f.v;
}

// B-fragment from transposed source: value(k,n) = base[n*ld + k]  (Q*K^T)
__device__ static inline v16bf load_frag_bt(const bf16* base, int ld) {
  const int lane  = threadIdx.x & 31;
  const int n     = lane & 15;
  const int khalf = (lane >> 4) << 3;
  FragU f;
#pragma unroll
  for (int g = 0; g < 8; ++g) {
    int k = ((g < 4) ? (2 * g) : (16 + 2 * (g - 4))) + khalf;
    f.u[g] = *(const unsigned int*)(base + n * ld + k);
  }
  return f.v;
}

__device__ static inline float reduce_max16(float v) {
#pragma unroll
  for (int off = 8; off > 0; off >>= 1) v = fmaxf(v, __shfl_xor(v, off, 16));
  return v;
}
__device__ static inline float reduce_sum16(float v) {
#pragma unroll
  for (int off = 8; off > 0; off >>= 1) v += __shfl_xor(v, off, 16);
  return v;
}

// ---------------------------------------------------------------
// fp32 -> bf16 conversion
// ---------------------------------------------------------------
__global__ void cvt_f32_bf16(const float* __restrict__ src, bf16* __restrict__ dst, int n) {
  int i = blockIdx.x * 256 + threadIdx.x;
  if (i < n) dst[i] = (bf16)src[i];
}

// ---------------------------------------------------------------
// Generic bf16 GEMM: C(MxN) = A(MxK,row) * B(KxN,row).
// Block tile 128x128, 8 waves of 32x64, K-step 32.
// TDM path: double-buffered LDS, wave0 DMAs tile t+1 while all compute tile t;
// TDM pad feature produces the +8-element padded LDS strides directly.
// ---------------------------------------------------------------
template <bool OUT_BF16>
__global__ __launch_bounds__(256) void gemm_bf16_wmma(
    const bf16* __restrict__ A, const bf16* __restrict__ Bm,
    void* __restrict__ Cout, int M, int N, int K) {
  constexpr int LDA = 40;                       // 32 + 8 pad  (80B rows)
  constexpr int LDB = 136;                      // 128 + 8 pad (272B rows)
  constexpr unsigned A_BYTES = 128 * LDA * 2;   // 10240 per buffer
  constexpr unsigned B_BYTES = 32 * LDB * 2;    // 8704 per buffer
  __shared__ bf16 smem[2 * 128 * LDA + 2 * 32 * LDB];   // 37.9 KB
  bf16* As = smem;                              // LDS byte off 0
  bf16* Bs = smem + 2 * 128 * LDA;              // LDS byte off 2*A_BYTES

  const int tid   = threadIdx.x;
  const int wave  = tid >> 5;
  const int lane  = tid & 31;
  const int mBase = blockIdx.y * 128;
  const int nBase = blockIdx.x * 128;
  const int wm    = (wave & 3) * 32;
  const int wn    = (wave >> 2) * 64;

  v8f acc[2][4];
  v8f z = {};
#pragma unroll
  for (int i = 0; i < 2; ++i)
#pragma unroll
    for (int j = 0; j < 4; ++j) acc[i][j] = z;

  const int T = K / 32;
#if USE_TDM
  if (wave == 0) {   // prologue: DMA tile 0 into buffer 0
    // A tile 32x128: rows of 16 DWORDs -> pad_interval=3 (16DW), pad_amount=3 (4DW=8 elems)
    tdm_load_2d(0u, A + (size_t)mBase * K, 32, 128, (unsigned)K, 1u, 3u, 3u);
    // B tile 128x32: rows of 64 DWORDs -> pad_interval=5 (64DW), pad_amount=3
    tdm_load_2d(2 * A_BYTES, Bm + nBase, 128, 32, (unsigned)N, 1u, 5u, 3u);
  }
  for (int t = 0; t < T; ++t) {
    const int buf = t & 1;
    if (wave == 0) {
      if (t + 1 < T) {
        const int k1 = (t + 1) * 32;
        tdm_load_2d((unsigned)((buf ^ 1) * A_BYTES),
                    A + (size_t)mBase * K + k1, 32, 128, (unsigned)K, 1u, 3u, 3u);
        tdm_load_2d(2 * A_BYTES + (unsigned)((buf ^ 1) * B_BYTES),
                    Bm + (size_t)k1 * N + nBase, 128, 32, (unsigned)N, 1u, 5u, 3u);
        __builtin_amdgcn_s_wait_tensorcnt(2);   // in-order: oldest 2 (tile t) done
      } else {
        __builtin_amdgcn_s_wait_tensorcnt(0);
      }
    }
    __syncthreads();
    const bf16* Ab = As + buf * 128 * LDA;
    const bf16* Bb = Bs + buf * 32 * LDB;
    v16bf aF[2], bF[4];
#pragma unroll
    for (int i = 0; i < 2; ++i) aF[i] = load_frag_a(Ab + (wm + i * 16) * LDA, LDA);
#pragma unroll
    for (int j = 0; j < 4; ++j) bF[j] = load_frag_b(Bb + wn + j * 16, LDB);
#pragma unroll
    for (int i = 0; i < 2; ++i)
#pragma unroll
      for (int j = 0; j < 4; ++j) acc[i][j] = wmma_bf16(aF[i], bF[j], acc[i][j]);
    __syncthreads();   // all waves done with buf before it is re-filled
  }
#else
  for (int t = 0; t < T; ++t) {
    const int k0 = t * 32;
#pragma unroll
    for (int it = 0; it < 2; ++it) {
      int c = tid + it * 256;
      int row = c >> 2, cc = c & 3;
      *(uint4*)(As + row * LDA + cc * 8) =
          *(const uint4*)(A + (size_t)(mBase + row) * K + k0 + cc * 8);
      int rb = c >> 4, cb = c & 15;
      *(uint4*)(Bs + rb * LDB + cb * 8) =
          *(const uint4*)(Bm + (size_t)(k0 + rb) * N + nBase + cb * 8);
    }
    if (k0 + 32 < K) {
      __builtin_prefetch(A + (size_t)(mBase + (tid >> 1)) * K + k0 + 32, 0, 0);
      __builtin_prefetch(Bm + (size_t)(k0 + 32 + (tid >> 4)) * N + nBase, 0, 0);
    }
    __syncthreads();
    v16bf aF[2], bF[4];
#pragma unroll
    for (int i = 0; i < 2; ++i) aF[i] = load_frag_a(As + (wm + i * 16) * LDA, LDA);
#pragma unroll
    for (int j = 0; j < 4; ++j) bF[j] = load_frag_b(Bs + wn + j * 16, LDB);
#pragma unroll
    for (int i = 0; i < 2; ++i)
#pragma unroll
      for (int j = 0; j < 4; ++j) acc[i][j] = wmma_bf16(aF[i], bF[j], acc[i][j]);
    __syncthreads();
  }
#endif

  const int nOff = lane & 15, mHalf = (lane >> 4) * 8;
#pragma unroll
  for (int i = 0; i < 2; ++i)
#pragma unroll
    for (int j = 0; j < 4; ++j)
#pragma unroll
      for (int g = 0; g < 8; ++g) {
        size_t row = (size_t)(mBase + wm + i * 16 + mHalf + g);
        size_t col = (size_t)(nBase + wn + j * 16 + nOff);
        if (OUT_BF16) ((bf16*)Cout)[row * N + col]  = (bf16)acc[i][j][g];
        else          ((float*)Cout)[row * N + col] = acc[i][j][g];
      }
}

// ---------------------------------------------------------------
// RoPE in-place on bf16 activations [B*S][heads*128]
// ---------------------------------------------------------------
__global__ void rope_bf16(bf16* __restrict__ t, int heads, int rowStride) {
  long idx = (long)blockIdx.x * 256 + threadIdx.x;
  int  i   = (int)(idx & 63);
  long tmp = idx >> 6;
  int  h   = (int)(tmp % heads);
  long bs  = tmp / heads;
  int  s   = (int)(bs % S_);
  float inv = __powf(10000.0f, -(float)i / 64.0f);
  float sn, cs;
  __sincosf((float)s * inv, &sn, &cs);
  bf16* p  = t + bs * rowStride + h * D_;
  float lo = (float)p[i], hi = (float)p[i + 64];
  p[i]      = (bf16)(lo * cs - hi * sn);
  p[i + 64] = (bf16)(hi * cs + lo * sn);
}

// ---------------------------------------------------------------
// Causal flash attention, GQA (n_rep=4), WMMA for QK^T and PV.
// Grid: (S/64, H, B). Block 256 = 8 waves.
// K/V tiles staged by TDM (one D# each), Q tile by vector loads.
// ---------------------------------------------------------------
__global__ __launch_bounds__(256) void attn_fwd_wmma(
    const bf16* __restrict__ Q, const bf16* __restrict__ Kg,
    const bf16* __restrict__ Vg, bf16* __restrict__ Og) {
  __shared__ bf16 smem[3 * 64 * 128 + 4 * 16 * 64];   // 56 KB
  bf16* Qs = smem;                  // LDS byte off 0
  bf16* Ks = smem + 64 * 128;       // byte off 16384
  bf16* Vs = smem + 2 * 64 * 128;   // byte off 32768
  bf16* Ps = smem + 3 * 64 * 128;   // byte off 49152

  const int tid  = threadIdx.x;
  const int lane = tid & 31;
  const int wave = tid >> 5;
  const int qt   = blockIdx.x;
  const int h    = blockIdx.y;
  const int b    = blockIdx.z;
  const int kvh  = h >> 2;
  const int qBase = qt * 64;
  const int rBlk  = wave & 3;
  const int dHalf = wave >> 2;
  const int nOff  = lane & 15, mHalf = (lane >> 4) * 8;
  const float scale = 0.08838834764831845f;   // 1/sqrt(128)

#pragma unroll
  for (int it = 0; it < 4; ++it) {
    int c = tid + it * 256;
    int row = c >> 4, cc = c & 15;
    *(uint4*)(Qs + row * 128 + cc * 8) =
        *(const uint4*)(Q + (size_t)(b * S_ + qBase + row) * E_ + h * D_ + cc * 8);
  }

  float mrow[8], lrow[8];
  v8f oacc[4];
  v8f z = {};
#pragma unroll
  for (int g = 0; g < 8; ++g) { mrow[g] = -3.402823466e38f; lrow[g] = 0.0f; }
#pragma unroll
  for (int j = 0; j < 4; ++j) oacc[j] = z;

  for (int kt = 0; kt <= qt; ++kt) {
    const int kBase = kt * 64;
    __syncthreads();   // prior iteration's K/V consumers done
#if USE_TDM
    if (wave == 0) {
      const size_t srcRow = (size_t)(b * S_ + kBase) * KV_ + kvh * D_;
      tdm_load_2d(16384u, Kg + srcRow, 128, 64, KV_, 0u, 0u, 0u);
      tdm_load_2d(32768u, Vg + srcRow, 128, 64, KV_, 0u, 0u, 0u);
      __builtin_amdgcn_s_wait_tensorcnt(0);
    }
#else
#pragma unroll
    for (int it = 0; it < 4; ++it) {
      int c = tid + it * 256;
      int row = c >> 4, cc = c & 15;
      size_t src = (size_t)(b * S_ + kBase + row) * KV_ + kvh * D_ + cc * 8;
      *(uint4*)(Ks + row * 128 + cc * 8) = *(const uint4*)(Kg + src);
      *(uint4*)(Vs + row * 128 + cc * 8) = *(const uint4*)(Vg + src);
    }
#endif
    __syncthreads();

    // ---- scores: S(16x64) = Q_rBlk(16x128) x K^T(128x64)
    v8f sAcc[4];
#pragma unroll
    for (int j = 0; j < 4; ++j) sAcc[j] = z;
#pragma unroll
    for (int kk = 0; kk < 4; ++kk) {
      v16bf aF = load_frag_a(Qs + rBlk * 16 * 128 + kk * 32, 128);
#pragma unroll
      for (int j = 0; j < 4; ++j) {
        v16bf bF = load_frag_bt(Ks + (j * 16) * 128 + kk * 32, 128);
        sAcc[j] = wmma_bf16(aF, bF, sAcc[j]);
      }
    }

    // ---- scale + causal mask + online softmax (16-lane shfl row stats)
    const bool diag = (kt == qt);
    float corr[8];
#pragma unroll
    for (int g = 0; g < 8; ++g) {
      const int row = qBase + rBlk * 16 + mHalf + g;
      float vmax = -3.402823466e38f;
#pragma unroll
      for (int j = 0; j < 4; ++j) {
        float s = sAcc[j][g] * scale;
        if (diag && (kBase + j * 16 + nOff) > row) s = -3.402823466e38f;
        sAcc[j][g] = s;
        vmax = fmaxf(vmax, s);
      }
      vmax = reduce_max16(vmax);
      float mn = fmaxf(mrow[g], vmax);
      corr[g]  = __expf(mrow[g] - mn);
      mrow[g]  = mn;
      float rs = 0.0f;
#pragma unroll
      for (int j = 0; j < 4; ++j) {
        float p = __expf(sAcc[j][g] - mn);
        sAcc[j][g] = p;
        rs += p;
      }
      lrow[g] = lrow[g] * corr[g] + reduce_sum16(rs);
    }
#pragma unroll
    for (int j = 0; j < 4; ++j)
#pragma unroll
      for (int g = 0; g < 8; ++g) oacc[j][g] *= corr[g];

    // ---- P (C-layout) -> LDS bf16 -> reload in A-layout for PV
    __syncthreads();
    if (wave < 4) {
      bf16* pp = Ps + rBlk * 16 * 64;
#pragma unroll
      for (int j = 0; j < 4; ++j)
#pragma unroll
        for (int g = 0; g < 8; ++g)
          pp[(mHalf + g) * 64 + j * 16 + nOff] = (bf16)sAcc[j][g];
    }
    __syncthreads();

    // ---- O += P(16x64) x V(64 x 64-slice)
#pragma unroll
    for (int kk = 0; kk < 2; ++kk) {
      v16bf aF = load_frag_a(Ps + rBlk * 16 * 64 + kk * 32, 64);
#pragma unroll
      for (int j = 0; j < 4; ++j) {
        v16bf bF = load_frag_b(Vs + kk * 32 * 128 + dHalf * 64 + j * 16, 128);
        oacc[j] = wmma_bf16(aF, bF, oacc[j]);
      }
    }
  }

#pragma unroll
  for (int j = 0; j < 4; ++j)
#pragma unroll
    for (int g = 0; g < 8; ++g) {
      float o = oacc[j][g] / lrow[g];
      size_t row = (size_t)(b * S_ + qBase + rBlk * 16 + mHalf + g);
      int    col = h * D_ + dHalf * 64 + j * 16 + nOff;
      Og[row * E_ + col] = (bf16)o;
    }
}

// ---------------------------------------------------------------
// launcher
// ---------------------------------------------------------------
extern "C" void kernel_launch(void* const* d_in, const int* in_sizes, int n_in,
                              void* d_out, int out_size, void* d_ws, size_t ws_size,
                              hipStream_t stream) {
  const float* x  = (const float*)d_in[0];
  const float* Wq = (const float*)d_in[1];
  const float* Wk = (const float*)d_in[2];
  const float* Wv = (const float*)d_in[3];
  const float* Wo = (const float*)d_in[4];

  char* ws = (char*)d_ws;
  size_t off = 0;
  auto carve = [&](size_t elems) { bf16* p = (bf16*)(ws + off); off += elems * sizeof(bf16); return p; };

  bf16* xb  = carve((size_t)M_ * E_);
  bf16* wqb = carve((size_t)E_ * E_);
  bf16* wkb = carve((size_t)E_ * KV_);
  bf16* wvb = carve((size_t)E_ * KV_);
  bf16* wob = carve((size_t)E_ * E_);
  bf16* qb  = carve((size_t)M_ * E_);
  bf16* kb  = carve((size_t)M_ * KV_);
  bf16* vb  = carve((size_t)M_ * KV_);
  bf16* ab  = carve((size_t)M_ * E_);

  cvt_f32_bf16<<<(M_ * E_) / 256, 256, 0, stream>>>(x,  xb,  M_ * E_);
  cvt_f32_bf16<<<(E_ * E_) / 256, 256, 0, stream>>>(Wq, wqb, E_ * E_);
  cvt_f32_bf16<<<(E_ * KV_) / 256, 256, 0, stream>>>(Wk, wkb, E_ * KV_);
  cvt_f32_bf16<<<(E_ * KV_) / 256, 256, 0, stream>>>(Wv, wvb, E_ * KV_);
  cvt_f32_bf16<<<(E_ * E_) / 256, 256, 0, stream>>>(Wo, wob, E_ * E_);

  gemm_bf16_wmma<true><<<dim3(E_ / 128,  M_ / 128), 256, 0, stream>>>(xb, wqb, qb, M_, E_,  E_);
  gemm_bf16_wmma<true><<<dim3(KV_ / 128, M_ / 128), 256, 0, stream>>>(xb, wkb, kb, M_, KV_, E_);
  gemm_bf16_wmma<true><<<dim3(KV_ / 128, M_ / 128), 256, 0, stream>>>(xb, wvb, vb, M_, KV_, E_);

  rope_bf16<<<((long)M_ * H_   * 64) / 256, 256, 0, stream>>>(qb, H_,   E_);
  rope_bf16<<<((long)M_ * KVH_ * 64) / 256, 256, 0, stream>>>(kb, KVH_, KV_);

  attn_fwd_wmma<<<dim3(S_ / 64, H_, B_), 256, 0, stream>>>(qb, kb, vb, ab);

  gemm_bf16_wmma<false><<<dim3(E_ / 128, M_ / 128), 256, 0, stream>>>(ab, wob, d_out, M_, E_, E_);
}
